// NativeGatedDeltaNet_80083960201377
// MI455X (gfx1250) — matrix operand order repaired and checked by reading, not compiled, not run on hardware
//
#include <hip/hip_runtime.h>
#include <hip/hip_bf16.h>

// ---------------- problem constants ----------------
#define B_   2
#define S_   2048
#define HID_ 2048
#define NK_  16
#define NV_  32
#define DK_  128
#define DV_  128
#define KC_  4
#define KEYD_ (NK_*DK_)          // 2048
#define VALD_ (NV_*DV_)          // 4096
#define CONVD_ (2*KEYD_+VALD_)   // 8192
#define MROWS_ (B_*S_)           // 4096
#define EPS_ 1e-6f

typedef __attribute__((ext_vector_type(16))) __bf16 v16bf;
typedef __attribute__((ext_vector_type(8)))  float  v8f;
typedef __attribute__((ext_vector_type(8)))  unsigned int v8u;

__device__ __forceinline__ unsigned short f32_to_bf16_bits(float f) {
  unsigned int x = __builtin_bit_cast(unsigned int, f);
  x += 0x7fffu + ((x >> 16) & 1u);     // round to nearest even
  return (unsigned short)(x >> 16);
}

// build a v16bf fragment from two 16-byte LDS chunks (documented CDNA5 layouts)
__device__ __forceinline__ v16bf make_frag(const unsigned short* lo_p,
                                           const unsigned short* hi_p) {
  uint4 lo = *(const uint4*)lo_p;
  uint4 hi = *(const uint4*)hi_p;
  v8u u;
  u[0] = lo.x; u[1] = lo.y; u[2] = lo.z; u[3] = lo.w;
  u[4] = hi.x; u[5] = hi.y; u[6] = hi.z; u[7] = hi.w;
  return __builtin_bit_cast(v16bf, u);
}

// ---------------- f32 -> bf16 cast ----------------
__global__ void cast_bf16_kernel(const float* __restrict__ src,
                                 unsigned short* __restrict__ dst, size_t n) {
  size_t i = (size_t)blockIdx.x * blockDim.x + threadIdx.x;
  size_t stride = (size_t)gridDim.x * blockDim.x;
  for (; i < n; i += stride) dst[i] = f32_to_bf16_bits(src[i]);
}

// ---------------- WMMA bf16 GEMM: C[M,N] = A[M,K] * Bw[N,K]^T ----------------
// 256 threads = 8 waves; block tile 128x128; K-step 32; double-buffered LDS.
// Row stride 40 halves (80 B, 16B-aligned) so fragments are two b128 LDS loads.
// sched_group_barrier pins: 18 DS-reads -> one wait -> 8 back-to-back WMMAs.
__global__ __launch_bounds__(256)
void gemm_bf16_wmma(const unsigned short* __restrict__ A,
                    const unsigned short* __restrict__ Bw,
                    float* __restrict__ C,
                    int M, int N, int Kdim) {
  __shared__ __attribute__((aligned(16))) unsigned short Alds[2][128][40];
  __shared__ __attribute__((aligned(16))) unsigned short Blds[2][128][40];

  const int tid  = threadIdx.x;
  const int lane = tid & 31;
  const int wave = tid >> 5;
  const int l16  = lane & 15;
  const int hi   = lane >> 4;

  const int bm = blockIdx.y * 128;
  const int bn = blockIdx.x * 128;

  const int lrow = tid >> 1;         // 0..127
  const int lch  = (tid & 1) * 16;   // half-offset 0 or 16

  v8f acc[8];
#pragma unroll
  for (int j = 0; j < 8; ++j) acc[j] = 0.f;

  const unsigned short* ag = A  + (size_t)(bm + lrow) * Kdim + lch;
  const unsigned short* bg = Bw + (size_t)(bn + lrow) * Kdim + lch;

  // ---- prologue: stage K-tile 0 into buffer 0 ----
  uint4 a0 = *(const uint4*)(ag);
  uint4 a1 = *(const uint4*)(ag + 8);
  uint4 b0 = *(const uint4*)(bg);
  uint4 b1 = *(const uint4*)(bg + 8);
  {
    uint4* ad = (uint4*)&Alds[0][lrow][lch];
    uint4* bd = (uint4*)&Blds[0][lrow][lch];
    ad[0] = a0; ad[1] = a1;
    bd[0] = b0; bd[1] = b1;
  }
  __syncthreads();

  const int nk = Kdim >> 5;
  const int arow = wave * 16 + l16;

  for (int i = 0; i < nk; ++i) {
    const int buf = i & 1;
    // issue global loads for the next K-tile (overlaps with compute below)
    if (i + 1 < nk) {
      const unsigned short* agn = ag + (size_t)(i + 1) * 32;
      const unsigned short* bgn = bg + (size_t)(i + 1) * 32;
      a0 = *(const uint4*)(agn);
      a1 = *(const uint4*)(agn + 8);
      b0 = *(const uint4*)(bgn);
      b1 = *(const uint4*)(bgn + 8);
      if (i + 2 < nk) {                    // warm L2 for the tile after next
        __builtin_prefetch(agn + 32, 0, 3);
        __builtin_prefetch(bgn + 32, 0, 3);
      }
    }

    // ---- A fragment (16x32 bf16): halves [hi*8,+8) and [16+hi*8,+8) ----
    v16bf afrag = make_frag(&Alds[buf][arow][hi * 8],
                            &Alds[buf][arow][16 + hi * 8]);

    // ---- all 8 B fragments live, then 8 back-to-back WMMAs ----
    v16bf bfr[8];
#pragma unroll
    for (int j = 0; j < 8; ++j) {
      const unsigned short* bp = &Blds[buf][j * 16 + l16][hi * 8];
      bfr[j] = make_frag(bp, bp + 16);
    }
#pragma unroll
    for (int j = 0; j < 8; ++j) {
      acc[j] = __builtin_amdgcn_wmma_f32_16x16x32_bf16(
          false, afrag, false, bfr[j], (short)0, acc[j], false, false);
    }
#if __has_builtin(__builtin_amdgcn_sched_group_barrier)
    // Pin schedule: 18 DS reads (A:2 + B:16), then 8 WMMA in one burst.
    __builtin_amdgcn_sched_group_barrier(0x100, 18, 0);  // DS read
    __builtin_amdgcn_sched_group_barrier(0x008, 8, 0);   // MFMA/WMMA
#endif

    // ---- stage next tile into the other buffer; one barrier per K-step ----
    if (i + 1 < nk) {
      uint4* ad = (uint4*)&Alds[buf ^ 1][lrow][lch];
      uint4* bd = (uint4*)&Blds[buf ^ 1][lrow][lch];
      ad[0] = a0; ad[1] = a1;
      bd[0] = b0; bd[1] = b1;
      __syncthreads();
    }
  }

  // ---- epilogue: C/D layout (VGPR r -> row r or r+8, lane%16 -> col) ----
#pragma unroll
  for (int j = 0; j < 8; ++j) {
    int n = bn + j * 16 + l16;
#pragma unroll
    for (int r = 0; r < 8; ++r) {
      int m = bm + wave * 16 + r + hi * 8;
      C[(size_t)m * N + n] = acc[j][r];
    }
  }
}

// ---------------- small projections: beta / g ----------------
__global__ __launch_bounds__(64)
void proj_bg_kernel(const float* __restrict__ hs,
                    const float* __restrict__ Wb,
                    const float* __restrict__ Wa,
                    const float* __restrict__ dt_bias,
                    const float* __restrict__ A_log,
                    float* __restrict__ beta,
                    float* __restrict__ g) {
  int m = blockIdx.x;            // 0..MROWS-1
  int h = blockIdx.y;            // 0..NV-1
  int wv = threadIdx.x >> 5;     // 0: beta, 1: g
  int lane = threadIdx.x & 31;
  const float* w = (wv == 0 ? Wb : Wa) + (size_t)h * HID_;
  const float* x = hs + (size_t)m * HID_;
  float acc = 0.f;
  for (int k = lane; k < HID_; k += 32) acc += x[k] * w[k];
#pragma unroll
  for (int off = 16; off > 0; off >>= 1) acc += __shfl_xor(acc, off);
  if (lane == 0) {
    if (wv == 0) {
      beta[(size_t)m * NV_ + h] = 1.f / (1.f + __expf(-acc));
    } else {
      float xdt = acc + dt_bias[h];
      float sp = (xdt > 20.f) ? xdt : log1pf(__expf(xdt));   // softplus
      g[(size_t)m * NV_ + h] = -__expf(A_log[h]) * sp;
    }
  }
}

// ---------------- causal depthwise conv(K=4) + SiLU + split + L2norm ----------------
__global__ __launch_bounds__(256)
void conv_silu_norm_kernel(const float* __restrict__ mixed,   // [MROWS, CONVD]
                           const float* __restrict__ conv_w,  // [CONVD, 4]
                           float* __restrict__ qbuf,          // [MROWS, KEYD]
                           float* __restrict__ kbuf,          // [MROWS, KEYD]
                           float* __restrict__ vbuf) {        // [MROWS, VALD]
  __shared__ float qk[2 * KEYD_];   // q,k channels post conv+silu (16 KB)
  int row = blockIdx.x;             // b*S + s
  int s = row & (S_ - 1);
  int tid = threadIdx.x;

  for (int c = tid; c < CONVD_; c += 256) {
    float acc = 0.f;
#pragma unroll
    for (int t = 0; t < KC_; ++t) {
      int sp = s - (KC_ - 1) + t;
      if (sp >= 0) acc += conv_w[c * KC_ + t] * mixed[(size_t)(row - (KC_-1) + t) * CONVD_ + c];
    }
    float val = acc / (1.f + __expf(-acc));   // silu
    if (c < 2 * KEYD_) qk[c] = val;
    else vbuf[(size_t)row * VALD_ + (c - 2 * KEYD_)] = val;
  }
  __syncthreads();

  // 32 chunks (16 q heads, 16 k heads) of DK=128, 8 threads per chunk
  int chunk = tid >> 3;
  int sub = tid & 7;
  float ss = 0.f;
  for (int i = sub; i < DK_; i += 8) { float t = qk[chunk * DK_ + i]; ss += t * t; }
  ss += __shfl_xor(ss, 1); ss += __shfl_xor(ss, 2); ss += __shfl_xor(ss, 4);
  float rn = rsqrtf(ss + EPS_);
  for (int i = sub; i < DK_; i += 8) {
    float val = qk[chunk * DK_ + i] * rn;
    if (chunk < NK_) qbuf[(size_t)row * KEYD_ + chunk * DK_ + i] = val;
    else             kbuf[(size_t)row * KEYD_ + (chunk - NK_) * DK_ + i] = val;
  }
}

// ---------------- gated delta-rule recurrence ----------------
// one block per (b,h); thread = (khalf, vcol); 64 state elems per thread in VGPRs.
__global__ __launch_bounds__(256)
void recurrence_kernel(const float* __restrict__ qbuf,   // [MROWS, NK, DK]
                       const float* __restrict__ kbuf,   // [MROWS, NK, DK]
                       const float* __restrict__ vbuf,   // [MROWS, NV, DV]
                       const float* __restrict__ gbuf,   // [MROWS, NV]
                       const float* __restrict__ bbuf,   // [MROWS, NV]
                       float* __restrict__ obuf) {       // [MROWS, NV, DV]
  __shared__ float sh_k[DK_], sh_q[DK_], sh_v[DV_];
  __shared__ float sh_part[2][DV_];
  __shared__ float sh_g, sh_b;

  int blk = blockIdx.x;
  int b = blk / NV_;
  int h = blk % NV_;
  int hk = h >> 1;               // GQA source head (rep = 2)
  int tid = threadIdx.x;
  int vcol = tid & (DV_ - 1);
  int khalf = tid >> 7;          // 0 or 1
  const float scale = 0.08838834764831845f;   // 128^-0.5

  float st[64];
#pragma unroll
  for (int i = 0; i < 64; ++i) st[i] = 0.f;

  for (int s = 0; s < S_; ++s) {
    size_t row = (size_t)b * S_ + s;
    if (tid < DK_) {
      sh_k[tid] = kbuf[row * KEYD_ + hk * DK_ + tid];
      sh_q[tid] = qbuf[row * KEYD_ + hk * DK_ + tid];
    } else {
      sh_v[tid - DK_] = vbuf[row * VALD_ + h * DV_ + (tid - DK_)];
    }
    if (tid == 0) { sh_g = gbuf[row * NV_ + h]; sh_b = bbuf[row * NV_ + h]; }
    __syncthreads();

    float decay = __expf(sh_g);
    float part = 0.f;
#pragma unroll
    for (int i = 0; i < 64; ++i) {
      float v = st[i] * decay;
      st[i] = v;
      part += sh_k[khalf * 64 + i] * v;
    }
    sh_part[khalf][vcol] = part;
    __syncthreads();

    float mem = sh_part[0][vcol] + sh_part[1][vcol];
    float delta = (sh_v[vcol] - mem) * sh_b;

    float opart = 0.f;
#pragma unroll
    for (int i = 0; i < 64; ++i) {
      float kk = sh_k[khalf * 64 + i];
      float qq = sh_q[khalf * 64 + i];
      st[i] += kk * delta;
      opart += qq * st[i];
    }
    sh_part[khalf][vcol] = opart;
    __syncthreads();

    if (khalf == 0)
      obuf[row * VALD_ + h * DV_ + vcol] = scale * (sh_part[0][vcol] + sh_part[1][vcol]);
    __syncthreads();
  }
}

// ---------------- gated RMSNorm + SiLU(z) gate -> bf16 ----------------
__global__ __launch_bounds__(128)
void gate_norm_kernel(const float* __restrict__ obuf,
                      const float* __restrict__ zbuf,
                      const float* __restrict__ norm_w,
                      unsigned short* __restrict__ og) {
  __shared__ float red[128];
  size_t idx = blockIdx.x;       // (b*S+s)*NV + h
  int d = threadIdx.x;
  float ov = obuf[idx * DV_ + d];
  red[d] = ov * ov;
  __syncthreads();
#pragma unroll
  for (int off = 64; off > 0; off >>= 1) {
    if (d < off) red[d] += red[d + off];
    __syncthreads();
  }
  float var = red[0] * (1.f / DV_);
  float on = ov * rsqrtf(var + EPS_) * norm_w[d];
  float zv = zbuf[idx * DV_ + d];
  float gated = on * (zv / (1.f + __expf(-zv)));
  og[idx * DV_ + d] = f32_to_bf16_bits(gated);
}

// ---------------- host side ----------------
extern "C" void kernel_launch(void* const* d_in, const int* in_sizes, int n_in,
                              void* d_out, int out_size, void* d_ws, size_t ws_size,
                              hipStream_t stream) {
  const float* hs      = (const float*)d_in[0];
  const float* W_qkv   = (const float*)d_in[1];
  const float* W_z     = (const float*)d_in[2];
  const float* W_b     = (const float*)d_in[3];
  const float* W_a     = (const float*)d_in[4];
  const float* conv_w  = (const float*)d_in[5];
  const float* norm_w  = (const float*)d_in[6];
  const float* W_out   = (const float*)d_in[7];
  const float* dt_bias = (const float*)d_in[8];
  const float* A_log   = (const float*)d_in[9];
  float* out = (float*)d_out;

  size_t off = 0;
  auto alloc = [&](size_t bytes) -> void* {
    void* p = (char*)d_ws + off;
    off += (bytes + 255) & ~(size_t)255;
    return p;
  };

  unsigned short* hsb   = (unsigned short*)alloc((size_t)MROWS_ * HID_ * 2);
  unsigned short* wqkvb = (unsigned short*)alloc((size_t)CONVD_ * HID_ * 2);
  unsigned short* wzb   = (unsigned short*)alloc((size_t)VALD_ * HID_ * 2);
  unsigned short* woutb = (unsigned short*)alloc((size_t)HID_ * VALD_ * 2);
  float* mixed = (float*)alloc((size_t)MROWS_ * CONVD_ * 4);
  float* zbuf  = (float*)alloc((size_t)MROWS_ * VALD_ * 4);
  float* qbuf  = (float*)alloc((size_t)MROWS_ * KEYD_ * 4);
  float* kbuf  = (float*)alloc((size_t)MROWS_ * KEYD_ * 4);
  float* vbuf  = (float*)alloc((size_t)MROWS_ * VALD_ * 4);
  float* beta  = (float*)alloc((size_t)MROWS_ * NV_ * 4);
  float* gbuf  = (float*)alloc((size_t)MROWS_ * NV_ * 4);
  float* obuf  = (float*)alloc((size_t)MROWS_ * VALD_ * 4);
  unsigned short* ogb = (unsigned short*)alloc((size_t)MROWS_ * VALD_ * 2);

  // 1) casts to bf16
  cast_bf16_kernel<<<2048, 256, 0, stream>>>(hs,    hsb,   (size_t)MROWS_ * HID_);
  cast_bf16_kernel<<<4096, 256, 0, stream>>>(W_qkv, wqkvb, (size_t)CONVD_ * HID_);
  cast_bf16_kernel<<<2048, 256, 0, stream>>>(W_z,   wzb,   (size_t)VALD_ * HID_);
  cast_bf16_kernel<<<2048, 256, 0, stream>>>(W_out, woutb, (size_t)HID_ * VALD_);

  // 2) big WMMA GEMMs: mixed = hs @ Wqkv^T, z = hs @ Wz^T
  gemm_bf16_wmma<<<dim3(CONVD_/128, MROWS_/128), 256, 0, stream>>>(
      hsb, wqkvb, mixed, MROWS_, CONVD_, HID_);
  gemm_bf16_wmma<<<dim3(VALD_/128, MROWS_/128), 256, 0, stream>>>(
      hsb, wzb, zbuf, MROWS_, VALD_, HID_);

  // 3) beta / g projections (f32, tiny)
  proj_bg_kernel<<<dim3(MROWS_, NV_), 64, 0, stream>>>(
      hs, W_b, W_a, dt_bias, A_log, beta, gbuf);

  // 4) conv + silu + split + l2norm
  conv_silu_norm_kernel<<<MROWS_, 256, 0, stream>>>(mixed, conv_w, qbuf, kbuf, vbuf);

  // 5) sequential gated delta recurrence (64 independent head instances)
  recurrence_kernel<<<B_ * NV_, 256, 0, stream>>>(qbuf, kbuf, vbuf, gbuf, beta, obuf);

  // 6) gated RMSNorm + silu(z) gate -> bf16
  gate_norm_kernel<<<MROWS_ * NV_, 128, 0, stream>>>(obuf, zbuf, norm_w, ogb);

  // 7) final WMMA GEMM: out = og @ Wout^T
  gemm_bf16_wmma<<<dim3(HID_/128, MROWS_/128), 256, 0, stream>>>(
      ogb, woutb, out, MROWS_, HID_, VALD_);
}
